// ProtowithCrossSimilarityScorer_80049600463390
// MI455X (gfx1250) — matrix operand order repaired and checked by reading, not compiled, not run on hardware
//
#include <hip/hip_runtime.h>

// ---------------------------------------------------------------------------
// MI455X reasoning: workload is HBM-bound on one 402MB read of test_reps
// (~17us @ 23.3 TB/s). All matrix math (~1.0 GFLOP after exploiting the
// one-hot sparsity of `support`) runs on v_wmma_f32_16x16x32_bf16; the 25MB
// `test` intermediate stays L2-resident (192MB L2) between kernels.
// ---------------------------------------------------------------------------

namespace {
constexpr int kB = 4, kS = 16, kTQ = 2048, kD = 768, kNT = 64;
constexpr int kNK = kS * kNT;           // 1024
constexpr int kMID1 = 204, kMID2 = 409;
}

typedef unsigned short u16;
typedef __attribute__((ext_vector_type(16))) __bf16   v16bf;
typedef __attribute__((ext_vector_type(8)))  float    v8f;
typedef __attribute__((ext_vector_type(4)))  float    f4;
typedef __attribute__((ext_vector_type(4)))  unsigned u32x4;
typedef __attribute__((ext_vector_type(2)))  unsigned u32x2;

__device__ __forceinline__ u16 f2bf(float f) {
  unsigned x = __float_as_uint(f);
  unsigned r = x + 0x7FFFu + ((x >> 16) & 1u);   // round-to-nearest-even
  return (u16)(r >> 16);
}

union BFrag { u32x4 u[2]; v16bf v; };

// ISA 7.12.2 16-bit A-matrix 16x32: lane<16 holds K {0..7, 16..23},
// lane>=16 holds K {8..15, 24..31} (two 16B chunks).
__device__ __forceinline__ v16bf load_a_frag(const u16* row, int k, int hi) {
  BFrag f;
  f.u[0] = *(const u32x4*)(row + k + 8 * hi);
  f.u[1] = *(const u32x4*)(row + k + 16 + 8 * hi);
  return f.v;
}
// B-matrix 32x16: lane<16 holds K {0..7, 16..23},
// lane>=16 holds K {8..15, 24..31} (two 16B chunks). N = lane%16.
__device__ __forceinline__ v16bf load_b_frag(const u16* row, int k, int hi) {
  BFrag f;
  f.u[0] = *(const u32x4*)(row + k + 8 * hi);
  f.u[1] = *(const u32x4*)(row + k + 16 + 8 * hi);
  return f.v;
}

// ---------------- K0: labels, tag counts, support -> bf16 -------------------
__global__ void k_prep(const float* __restrict__ tgt, const float* __restrict__ sup,
                       int* __restrict__ label, float* __restrict__ tag_count,
                       u16* __restrict__ sup_bf) {
  int tid = blockIdx.x * blockDim.x + threadIdx.x;
  for (int i = tid; i < kB * kS * kD; i += gridDim.x * blockDim.x)
    sup_bf[i] = f2bf(sup[i]);
  if (blockIdx.x == 0) {
    int t = threadIdx.x;
    if (t < kB * kNT) {                       // tag_count[b,n] = sum_s onehot
      int b = t / kNT, n = t % kNT;
      float c = 0.f;
      for (int s = 0; s < kS; ++s) c += tgt[(b * kS + s) * kNT + n];
      tag_count[t] = c;
    }
    if (t < kB * kS) {                        // label[b,s] = argmax onehot
      const float* r = tgt + t * kNT;
      int best = 0; float bv = r[0];
      for (int n = 1; n < kNT; ++n) { float v = r[n]; if (v > bv) { bv = v; best = n; } }
      label[t] = best;
    }
  }
}

// ---------------- K1: test = mean_S(test_reps); fp32 + bf16 -----------------
__global__ void k_mean(const float* __restrict__ test_reps,
                       float* __restrict__ test_f32, u16* __restrict__ test_bf) {
  constexpr int QD4 = kTQ * kD / 4;
  int idx = blockIdx.x * blockDim.x + threadIdx.x;
  if (idx >= kB * QD4) return;
  int b = idx / QD4, qd4 = idx % QD4;
  const f4* src = (const f4*)test_reps + (size_t)b * kS * QD4 + qd4;
  float ax = 0.f, ay = 0.f, az = 0.f, aw = 0.f;
  for (int s = 0; s < kS; ++s) {
    f4 v = __builtin_nontemporal_load(src + (size_t)s * QD4);  // stream, don't pollute L2
    ax += v.x; ay += v.y; az += v.z; aw += v.w;
  }
  const float inv = 1.0f / (float)kS;
  f4 o; o.x = ax * inv; o.y = ay * inv; o.z = az * inv; o.w = aw * inv;
  ((f4*)test_f32)[idx] = o;
  u32x2 pk;
  pk.x = (unsigned)f2bf(o.x) | ((unsigned)f2bf(o.y) << 16);
  pk.y = (unsigned)f2bf(o.z) | ((unsigned)f2bf(o.w) << 16);
  ((u32x2*)test_bf)[idx] = pk;
}

// ---------------- K2: Wnz[b,s,q] = sup[b,s]·test[b,q]  (WMMA) ---------------
__global__ __launch_bounds__(32) void k_wnz(const u16* __restrict__ sup_bf,
                                            const u16* __restrict__ test_bf,
                                            float* __restrict__ Wnz) {
  int tile = blockIdx.x;
  int b = tile / (kTQ / 16);
  int q0 = (tile % (kTQ / 16)) * 16;
  int lane = threadIdx.x, row = lane & 15, hi = lane >> 4;
  const u16* arow = sup_bf + (size_t)(b * kS + row) * kD;           // A: 16 support rows
  const u16* brow = test_bf + (size_t)(b * kTQ + q0 + row) * kD;    // B^T: 16 query rows
  v8f c = {0.f, 0.f, 0.f, 0.f, 0.f, 0.f, 0.f, 0.f};
  for (int k = 0; k < kD; k += 32) {
    v16bf a = load_a_frag(arow, k, hi);
    v16bf bb = load_b_frag(brow, k, hi);
    c = __builtin_amdgcn_wmma_f32_16x16x32_bf16(false, a, false, bb, (short)0, c, false, false);
  }
  float* out = Wnz + (size_t)b * kS * kTQ + q0 + row;  // D: M=r+8*hi (s), N=row (q)
#pragma unroll
  for (int r = 0; r < 8; ++r) out[(size_t)(r + 8 * hi) * kTQ] = c[r];
}

// ---------------- K3a: per-support-row pools over q -------------------------
__global__ void k_rowpool(const float* __restrict__ Wnz,
                          float* __restrict__ rowmax, float* __restrict__ rowmean) {
  __shared__ float smx[256], ssm[256];
  int bs = blockIdx.x, t = threadIdx.x;
  const float* row = Wnz + (size_t)bs * kTQ;
  float mx = -3.4e38f, sm = 0.f;
  for (int q = t; q < kTQ; q += 256) { float v = row[q]; mx = fmaxf(mx, v); sm += v; }
  smx[t] = mx; ssm[t] = sm; __syncthreads();
  for (int o = 128; o > 0; o >>= 1) {
    if (t < o) { smx[t] = fmaxf(smx[t], smx[t + o]); ssm[t] += ssm[t + o]; }
    __syncthreads();
  }
  if (t == 0) { rowmax[bs] = smx[0]; rowmean[bs] = ssm[0] / (float)kTQ; }
}

// ---------------- K3b: per-query pools over all NK rows (zeros incl.) -------
__global__ void k_colpool(const float* __restrict__ Wnz,
                          float* __restrict__ colmax, float* __restrict__ colmean) {
  int idx = blockIdx.x * blockDim.x + threadIdx.x;
  if (idx >= kB * kTQ) return;
  int b = idx / kTQ, q = idx % kTQ;
  float mx = 0.f, sm = 0.f;                 // zero rows contribute 0 to max
  for (int s = 0; s < kS; ++s) {
    float v = Wnz[(size_t)(b * kS + s) * kTQ + q];
    mx = fmaxf(mx, v); sm += v;
  }
  colmax[idx] = mx;
  colmean[idx] = sm / (float)kNK;           // mean over full 1024 rows
}

// ---------------- K4: branch-1 MLPs + fu + softmax(+1) -> sscale ------------
__global__ void k_branch1(const float* __restrict__ rowmax, const float* __restrict__ rowmean,
                          const int* __restrict__ label,
                          const float* __restrict__ w1, const float* __restrict__ b1,
                          const float* __restrict__ w2, const float* __restrict__ b2,
                          const float* __restrict__ fw1, const float* __restrict__ fb1,
                          const float* __restrict__ fw2, const float* __restrict__ fb2,
                          float* __restrict__ sscale) {
  __shared__ float hm[kMID1], ha[kMID1];
  __shared__ float fused[kNK];
  __shared__ float red[256];
  __shared__ int nzidx[kS];
  int b = blockIdx.x, t = threadIdx.x;
  if (t < kS) nzidx[t] = t * kNT + label[b * kS + t];
  __syncthreads();
  // hidden layer: x has only 16 nonzero entries
  for (int j = t; j < kMID1; j += 256) {
    float am = b1[j], aa = b1[j];
    for (int s = 0; s < kS; ++s) {
      float w = w1[(size_t)nzidx[s] * kMID1 + j];
      am += rowmax[b * kS + s] * w;
      aa += rowmean[b * kS + s] * w;
    }
    hm[j] = fmaxf(am, 0.f); ha[j] = fmaxf(aa, 0.f);
  }
  __syncthreads();
  // output layer + fu for all NK rows (needed for softmax denominator)
  float lmax = -3.4e38f;
  for (int n = t; n < kNK; n += 256) {
    float om = b2[n], oa = b2[n];
    for (int j = 0; j < kMID1; ++j) {
      float w = w2[(size_t)j * kNK + n];
      om += hm[j] * w; oa += ha[j] * w;
    }
    float h = fmaxf(oa * fw1[0] + om * fw1[1] + fb1[0], 0.f);  // concat([mean,max])
    float f = h * fw2[0] + fb2[0];
    fused[n] = f; lmax = fmaxf(lmax, f);
  }
  red[t] = lmax; __syncthreads();
  for (int o = 128; o > 0; o >>= 1) { if (t < o) red[t] = fmaxf(red[t], red[t + o]); __syncthreads(); }
  float mx = red[0]; __syncthreads();
  float lsum = 0.f;
  for (int n = t; n < kNK; n += 256) lsum += expf(fused[n] - mx);
  red[t] = lsum; __syncthreads();
  for (int o = 128; o > 0; o >>= 1) { if (t < o) red[t] += red[t + o]; __syncthreads(); }
  float sum = red[0]; __syncthreads();
  if (t < kS) sscale[b * kS + t] = expf(fused[nzidx[t]] - mx) / sum + 1.0f;
}

// ---------------- K5a: branch-2 hidden layer (2048 -> 409), relu ------------
__global__ void k_b2_h(const float* __restrict__ colmax, const float* __restrict__ colmean,
                       const float* __restrict__ w1, const float* __restrict__ b1,
                       float* __restrict__ h2m, float* __restrict__ h2a) {
  int idx = blockIdx.x * blockDim.x + threadIdx.x;
  if (idx >= kB * kMID2) return;
  int b = idx / kMID2, j = idx % kMID2;
  float am = b1[j], aa = b1[j];
  for (int q = 0; q < kTQ; ++q) {
    float w = w1[(size_t)q * kMID2 + j];     // coalesced across j
    am += colmax[b * kTQ + q] * w;
    aa += colmean[b * kTQ + q] * w;
  }
  h2m[idx] = fmaxf(am, 0.f); h2a[idx] = fmaxf(aa, 0.f);
}

// ---------------- K5b: branch-2 output layer (409 -> 2048) + fu -------------
__global__ void k_b2_out(const float* __restrict__ h2m, const float* __restrict__ h2a,
                         const float* __restrict__ w2, const float* __restrict__ b2,
                         const float* __restrict__ fw1, const float* __restrict__ fb1,
                         const float* __restrict__ fw2, const float* __restrict__ fb2,
                         float* __restrict__ fused2) {
  int idx = blockIdx.x * blockDim.x + threadIdx.x;
  if (idx >= kB * kTQ) return;
  int b = idx / kTQ, q = idx % kTQ;
  float om = b2[q], oa = b2[q];
  for (int j = 0; j < kMID2; ++j) {
    float w = w2[(size_t)j * kTQ + q];       // coalesced across q
    om += h2m[b * kMID2 + j] * w;
    oa += h2a[b * kMID2 + j] * w;
  }
  float h = fmaxf(oa * fw1[0] + om * fw1[1] + fb1[0], 0.f);
  fused2[idx] = h * fw2[0] + fb2[0];
}

// ---------------- K5c: softmax over q (+1) -> qscale ------------------------
__global__ void k_qsoftmax(const float* __restrict__ fused2, float* __restrict__ qscale) {
  __shared__ float red[256];
  int b = blockIdx.x, t = threadIdx.x;
  const float* f = fused2 + (size_t)b * kTQ;
  float lmax = -3.4e38f;
  for (int q = t; q < kTQ; q += 256) lmax = fmaxf(lmax, f[q]);
  red[t] = lmax; __syncthreads();
  for (int o = 128; o > 0; o >>= 1) { if (t < o) red[t] = fmaxf(red[t], red[t + o]); __syncthreads(); }
  float mx = red[0]; __syncthreads();
  float lsum = 0.f;
  for (int q = t; q < kTQ; q += 256) lsum += expf(f[q] - mx);
  red[t] = lsum; __syncthreads();
  for (int o = 128; o > 0; o >>= 1) { if (t < o) red[t] += red[t + o]; __syncthreads(); }
  float sum = red[0]; __syncthreads();
  for (int q = t; q < kTQ; q += 256)
    qscale[b * kTQ + q] = expf(f[q] - mx) / sum + 1.0f;
}

// ---------------- K6: prototypes (fp32 out + bf16 copy) ---------------------
__global__ void k_proto(const float* __restrict__ sup, const int* __restrict__ label,
                        const float* __restrict__ sscale, const float* __restrict__ tag_count,
                        float* __restrict__ proto_out, u16* __restrict__ proto_bf) {
  int idx = blockIdx.x * blockDim.x + threadIdx.x;
  if (idx >= kB * kNT * kD) return;
  int b = idx / (kNT * kD), nd = idx % (kNT * kD);
  int n = nd / kD, d = nd % kD;
  float acc = 0.f;
  for (int s = 0; s < kS; ++s) {
    float m = (label[b * kS + s] == n) ? sscale[b * kS + s] : 0.0f;
    acc += m * sup[(size_t)(b * kS + s) * kD + d];
  }
  float pv = acc / (tag_count[b * kNT + n] + 1e-4f);
  proto_out[idx] = pv;
  proto_bf[idx] = f2bf(pv);
}

// ---------------- K7: qry_w = qscale * test -> bf16 -------------------------
__global__ void k_qw(const float* __restrict__ qscale, const float* __restrict__ test_f32,
                     u16* __restrict__ qw_bf) {
  constexpr int QD4 = kTQ * kD / 4;
  int idx = blockIdx.x * blockDim.x + threadIdx.x;
  if (idx >= kB * QD4) return;
  int bq = idx / (kD / 4);
  f4 v = ((const f4*)test_f32)[idx];
  float s = qscale[bq];
  u32x2 pk;
  pk.x = (unsigned)f2bf(v.x * s) | ((unsigned)f2bf(v.y * s) << 16);
  pk.y = (unsigned)f2bf(v.z * s) | ((unsigned)f2bf(v.w * s) << 16);
  ((u32x2*)qw_bf)[idx] = pk;
}

// ---------------- K8: sim = qry_w @ proto^T  (WMMA) -------------------------
__global__ __launch_bounds__(32) void k_sim(const u16* __restrict__ qw_bf,
                                            const u16* __restrict__ proto_bf,
                                            float* __restrict__ sim) {
  constexpr int TQ_T = kTQ / 16, NT_T = kNT / 16;
  int bid = blockIdx.x;
  int b = bid / (TQ_T * NT_T);
  int r = bid % (TQ_T * NT_T);
  int q0 = (r / NT_T) * 16, n0 = (r % NT_T) * 16;
  int lane = threadIdx.x, row = lane & 15, hi = lane >> 4;
  const u16* arow = qw_bf + (size_t)(b * kTQ + q0 + row) * kD;     // A: 16 query rows
  const u16* brow = proto_bf + (size_t)(b * kNT + n0 + row) * kD;  // B^T: 16 proto rows
  v8f c = {0.f, 0.f, 0.f, 0.f, 0.f, 0.f, 0.f, 0.f};
  for (int k = 0; k < kD; k += 32) {
    v16bf a = load_a_frag(arow, k, hi);
    v16bf bb = load_b_frag(brow, k, hi);
    c = __builtin_amdgcn_wmma_f32_16x16x32_bf16(false, a, false, bb, (short)0, c, false, false);
  }
  float* out = sim + ((size_t)(b * kTQ + q0)) * kNT + n0 + row;    // M->q, N->n
#pragma unroll
  for (int r2 = 0; r2 < 8; ++r2) out[(size_t)(r2 + 8 * hi) * kNT] = c[r2];
}

// ---------------------------------------------------------------------------
extern "C" void kernel_launch(void* const* d_in, const int* in_sizes, int n_in,
                              void* d_out, int out_size, void* d_ws, size_t ws_size,
                              hipStream_t stream) {
  (void)in_sizes; (void)n_in; (void)out_size; (void)ws_size;
  const float* test_reps       = (const float*)d_in[0];
  const float* support_reps    = (const float*)d_in[1];
  // d_in[2], d_in[3]: masks, unused by the reference
  const float* support_targets = (const float*)d_in[4];
  const float* l1_w1 = (const float*)d_in[5],  *l1_b1 = (const float*)d_in[6];
  const float* l1_w2 = (const float*)d_in[7],  *l1_b2 = (const float*)d_in[8];
  const float* l2_w1 = (const float*)d_in[9],  *l2_b1 = (const float*)d_in[10];
  const float* l2_w2 = (const float*)d_in[11], *l2_b2 = (const float*)d_in[12];
  const float* fu_w1 = (const float*)d_in[13], *fu_b1 = (const float*)d_in[14];
  const float* fu_w2 = (const float*)d_in[15], *fu_b2 = (const float*)d_in[16];

  float* sim_out   = (float*)d_out;                       // (B,TQ,NT)
  float* proto_out = sim_out + (size_t)kB * kTQ * kNT;    // (B,NT,D)

  char* cur = (char*)d_ws;
  auto carve = [&](size_t bytes) -> void* {
    void* r = (void*)cur;
    cur += (bytes + 255) & ~(size_t)255;
    return r;
  };
  float* test_f32  = (float*)carve(sizeof(float) * kB * kTQ * kD);
  u16*   test_bf   = (u16*)  carve(sizeof(u16)   * kB * kTQ * kD);
  u16*   qw_bf     = (u16*)  carve(sizeof(u16)   * kB * kTQ * kD);
  u16*   sup_bf    = (u16*)  carve(sizeof(u16)   * kB * kS * kD);
  u16*   proto_bf  = (u16*)  carve(sizeof(u16)   * kB * kNT * kD);
  float* Wnz       = (float*)carve(sizeof(float) * kB * kS * kTQ);
  int*   label     = (int*)  carve(sizeof(int)   * kB * kS);
  float* tag_count = (float*)carve(sizeof(float) * kB * kNT);
  float* rowmax    = (float*)carve(sizeof(float) * kB * kS);
  float* rowmean   = (float*)carve(sizeof(float) * kB * kS);
  float* colmax    = (float*)carve(sizeof(float) * kB * kTQ);
  float* colmean   = (float*)carve(sizeof(float) * kB * kTQ);
  float* h2m       = (float*)carve(sizeof(float) * kB * kMID2);
  float* h2a       = (float*)carve(sizeof(float) * kB * kMID2);
  float* fused2    = (float*)carve(sizeof(float) * kB * kTQ);
  float* sscale    = (float*)carve(sizeof(float) * kB * kS);
  float* qscale    = (float*)carve(sizeof(float) * kB * kTQ);

  k_prep<<<192, 256, 0, stream>>>(support_targets, support_reps, label, tag_count, sup_bf);
  k_mean<<<(kB * kTQ * kD / 4 + 255) / 256, 256, 0, stream>>>(test_reps, test_f32, test_bf);
  k_wnz<<<kB * (kTQ / 16), 32, 0, stream>>>(sup_bf, test_bf, Wnz);
  k_rowpool<<<kB * kS, 256, 0, stream>>>(Wnz, rowmax, rowmean);
  k_colpool<<<(kB * kTQ + 255) / 256, 256, 0, stream>>>(Wnz, colmax, colmean);
  k_branch1<<<kB, 256, 0, stream>>>(rowmax, rowmean, label, l1_w1, l1_b1, l1_w2, l1_b2,
                                    fu_w1, fu_b1, fu_w2, fu_b2, sscale);
  k_b2_h<<<(kB * kMID2 + 255) / 256, 256, 0, stream>>>(colmax, colmean, l2_w1, l2_b1, h2m, h2a);
  k_b2_out<<<(kB * kTQ + 255) / 256, 256, 0, stream>>>(h2m, h2a, l2_w2, l2_b2,
                                                       fu_w1, fu_b1, fu_w2, fu_b2, fused2);
  k_qsoftmax<<<kB, 256, 0, stream>>>(fused2, qscale);
  k_proto<<<(kB * kNT * kD + 255) / 256, 256, 0, stream>>>(support_reps, label, sscale, tag_count,
                                                           proto_out, proto_bf);
  k_qw<<<(kB * kTQ * kD / 4 + 255) / 256, 256, 0, stream>>>(qscale, test_f32, qw_bf);
  k_sim<<<kB * (kTQ / 16) * (kNT / 16), 32, 0, stream>>>(qw_bf, proto_bf, sim_out);
}
